// EQCNN_cls_85383949844895
// MI455X (gfx1250) — matrix operand order repaired and checked by reading, not compiled, not run on hardware
//
#include <hip/hip_runtime.h>

#define EPSV    1e-6f
#define BN_EPSV 1e-5f
#define KNN     20
#define SLOPEV  0.2f

typedef __attribute__((ext_vector_type(16))) _Float16 v16h;
typedef __attribute__((ext_vector_type(8)))  _Float16 v8h;
typedef __attribute__((ext_vector_type(8)))  float    v8f;

// ---------------------------------------------------------------------------
// WMMA GEMM, branch-free inner loop.
//   C[M x Ncol] = A * B (+ optional bias per column)
//   A  : [M][Apad]    f16, K-contiguous rows, zero-padded to KP (KP % 32 == 0)
//   BT : [Ncol][Bpad] f16, K-contiguous rows of B-transposed, zero-padded
// OOB rows/cols are clamped to row/col 0 for loads (their products only land
// in C entries that are never stored), so the k-loop has no guards at all.
// Each wave computes a 32x32 C macro-tile (2x2 WMMA tiles, fragments reused).
// Wave-uniform early exit keeps EXEC all-ones for V_WMMA.
// ---------------------------------------------------------------------------
template <typename OutT>
__global__ void gemm_wmma_kernel(const _Float16* __restrict__ A,
                                 const _Float16* __restrict__ BT,
                                 OutT* __restrict__ C,
                                 int M, int Ncol, int KP,
                                 long Apad, long Bpad, long Crs,
                                 const float* __restrict__ bias) {
    int wave = blockIdx.x * (blockDim.x >> 5) + (threadIdx.x >> 5);
    int tilesM2 = (M + 31) >> 5;
    int tilesN2 = (Ncol + 31) >> 5;
    if (wave >= tilesM2 * tilesN2) return;        // uniform per wave
    int tm = wave / tilesN2;
    int tn = wave % tilesN2;
    int lane = threadIdx.x & 31;
    int l  = lane & 15;
    int hi = lane >> 4;
    int arow0 = tm * 32 + l;
    int arow1 = arow0 + 16;
    int bcol0 = tn * 32 + l;
    int bcol1 = bcol0 + 16;
    const _Float16* Ap0 = A  + (long)(arow0 < M    ? arow0 : 0) * Apad + hi * 8;
    const _Float16* Ap1 = A  + (long)(arow1 < M    ? arow1 : 0) * Apad + hi * 8;
    const _Float16* Bp0 = BT + (long)(bcol0 < Ncol ? bcol0 : 0) * Bpad + hi * 8;
    const _Float16* Bp1 = BT + (long)(bcol1 < Ncol ? bcol1 : 0) * Bpad + hi * 8;
    v8f acc00 = {}, acc01 = {}, acc10 = {}, acc11 = {};
    for (int kk = 0; kk < KP; kk += 32) {
        __builtin_prefetch((const void*)(Bp0 + kk + 128), 0, 0);
        v8h a0lo = *(const v8h*)(Ap0 + kk);
        v8h a0hi = *(const v8h*)(Ap0 + kk + 16);
        v8h a1lo = *(const v8h*)(Ap1 + kk);
        v8h a1hi = *(const v8h*)(Ap1 + kk + 16);
        v8h b0lo = *(const v8h*)(Bp0 + kk);
        v8h b0hi = *(const v8h*)(Bp0 + kk + 16);
        v8h b1lo = *(const v8h*)(Bp1 + kk);
        v8h b1hi = *(const v8h*)(Bp1 + kk + 16);
        v16h av0 = __builtin_shufflevector(a0lo, a0hi, 0,1,2,3,4,5,6,7,8,9,10,11,12,13,14,15);
        v16h av1 = __builtin_shufflevector(a1lo, a1hi, 0,1,2,3,4,5,6,7,8,9,10,11,12,13,14,15);
        v16h bv0 = __builtin_shufflevector(b0lo, b0hi, 0,1,2,3,4,5,6,7,8,9,10,11,12,13,14,15);
        v16h bv1 = __builtin_shufflevector(b1lo, b1hi, 0,1,2,3,4,5,6,7,8,9,10,11,12,13,14,15);
        acc00 = __builtin_amdgcn_wmma_f32_16x16x32_f16(false, av0, false, bv0, (short)0, acc00, false, false);
        acc01 = __builtin_amdgcn_wmma_f32_16x16x32_f16(false, av0, false, bv1, (short)0, acc01, false, false);
        acc10 = __builtin_amdgcn_wmma_f32_16x16x32_f16(false, av1, false, bv0, (short)0, acc10, false, false);
        acc11 = __builtin_amdgcn_wmma_f32_16x16x32_f16(false, av1, false, bv1, (short)0, acc11, false, false);
    }
    float bv0 = (bias && bcol0 < Ncol) ? bias[bcol0] : 0.0f;
    float bv1 = (bias && bcol1 < Ncol) ? bias[bcol1] : 0.0f;
#pragma unroll
    for (int r = 0; r < 8; ++r) {                 // C: VGPR r -> row r + hi*8
        int row0 = tm * 32 + hi * 8 + r;
        int row1 = row0 + 16;
        if (row0 < M) {
            if (bcol0 < Ncol) C[(long)row0 * Crs + bcol0] = (OutT)(acc00[r] + bv0);
            if (bcol1 < Ncol) C[(long)row0 * Crs + bcol1] = (OutT)(acc01[r] + bv1);
        }
        if (row1 < M) {
            if (bcol0 < Ncol) C[(long)row1 * Crs + bcol0] = (OutT)(acc10[r] + bv0);
            if (bcol1 < Ncol) C[(long)row1 * Crs + bcol1] = (OutT)(acc11[r] + bv1);
        }
    }
}

__global__ void cast_f32_f16_kernel(const float* __restrict__ s,
                                    _Float16* __restrict__ d, long n) {
    for (long t = blockIdx.x * (long)blockDim.x + threadIdx.x; t < n;
         t += (long)gridDim.x * blockDim.x)
        d[t] = (_Float16)s[t];
}

// dst[r][k] = (k < K) ? src[r][k] : 0   with dst rows padded to KP.
__global__ void cast_pad_kernel(const float* __restrict__ s,
                                _Float16* __restrict__ d,
                                int R, int K, int KP) {
    long n = (long)R * KP;
    for (long t = blockIdx.x * (long)blockDim.x + threadIdx.x; t < n;
         t += (long)gridDim.x * blockDim.x) {
        int k = (int)(t % KP);
        long r = t / KP;
        d[t] = (k < K) ? (_Float16)s[r * K + k] : (_Float16)0.0f;
    }
}

__global__ void zero_f32_kernel(float* __restrict__ p, long n) {
    for (long t = blockIdx.x * (long)blockDim.x + threadIdx.x; t < n;
         t += (long)gridDim.x * blockDim.x)
        p[t] = 0.0f;
}

__global__ void zero_f16_kernel(_Float16* __restrict__ p, long n) {
    for (long t = blockIdx.x * (long)blockDim.x + threadIdx.x; t < n;
         t += (long)gridDim.x * blockDim.x)
        p[t] = (_Float16)0.0f;
}

// xx[n] = sum_d xf[d*N+n]^2   (xf = f32 [D][N])
__global__ void sqnorm_kernel(const float* __restrict__ xf, float* __restrict__ xx,
                              int D, int N) {
    int n = blockIdx.x * blockDim.x + threadIdx.x;
    if (n >= N) return;
    float s = 0.0f;
    for (int d = 0; d < D; ++d) {
        float v = xf[(long)d * N + n];
        s += v * v;
    }
    xx[n] = s;
}

// xfT[n][dp] = (dp < D) ? xcur[dp*N + n] : 0  -- K-contiguous Gram operand.
__global__ void build_xfT_kernel(const float* __restrict__ xcur,
                                 _Float16* __restrict__ xfT,
                                 int D, int N, int DP) {
    long n = (long)N * DP;
    for (long t = blockIdx.x * (long)blockDim.x + threadIdx.x; t < n;
         t += (long)gridDim.x * blockDim.x) {
        int dp = (int)(t % DP);
        long nn = t / DP;
        xfT[t] = (dp < D) ? (_Float16)xcur[(long)dp * N + nn] : (_Float16)0.0f;
    }
}

// Row-wise top-20 of pdist = 2*gram - xx_i - xx_j; fully-unrolled insertion
// keeps vals/ids in registers.
__global__ void topk_kernel(const float* __restrict__ gram,
                            const float* __restrict__ xx,
                            int* __restrict__ idx, int N) {
    int i = blockIdx.x * blockDim.x + threadIdx.x;
    if (i >= N) return;
    float vals[KNN];
    int   ids[KNN];
#pragma unroll
    for (int t = 0; t < KNN; ++t) { vals[t] = -3.0e38f; ids[t] = 0; }
    float xi = xx[i];
    const float* g = gram + (long)i * N;
    for (int j = 0; j < N; ++j) {
        float cv = 2.0f * g[j] - xi - xx[j];
        int   ci = j;
        if (cv > vals[KNN - 1]) {
#pragma unroll
            for (int t = 0; t < KNN; ++t) {
                if (cv > vals[t]) {
                    float tv = vals[t]; int ti = ids[t];
                    vals[t] = cv; ids[t] = ci;
                    cv = tv; ci = ti;
                }
            }
        }
    }
#pragma unroll
    for (int t = 0; t < KNN; ++t) idx[i * KNN + t] = ids[t];
}

// Edge features written TRANSPOSED: featT[m][KP], m = a*N*K + n*K + k,
// channel c2: [0,C) = neighbor - center, [C,2C) = center. Pad cols stay 0.
__global__ void gather_edge_kernel(const float* __restrict__ xcur,
                                   const int* __restrict__ idx,
                                   _Float16* __restrict__ featT,
                                   int C, int N, int Kk, int KP) {
    long total = (long)2 * C * 3 * N * Kk;
    int C2 = 2 * C;
    for (long t = blockIdx.x * (long)blockDim.x + threadIdx.x; t < total;
         t += (long)gridDim.x * blockDim.x) {
        int c2 = (int)(t % C2);
        long m = t / C2;
        int k  = (int)(m % Kk);
        long r = m / Kk;
        int n  = (int)(r % N);
        int a  = (int)(r / N);
        int c  = (c2 < C) ? c2 : c2 - C;
        float ctr = xcur[((long)c * 3 + a) * N + n];
        float v;
        if (c2 < C) {
            int nb = idx[n * Kk + k];
            v = xcur[((long)c * 3 + a) * N + nb] - ctr;
        } else {
            v = ctr;
        }
        featT[m * KP + c2] = (_Float16)v;
    }
}

// VN-BN statistics: per channel, LDS-reduced sum/sumsq of ||p||+EPS, one
// atomicAdd per block; accumulated across per-batch launches.
__global__ void bn_stats_kernel(const _Float16* __restrict__ p,
                                float* __restrict__ sum,
                                float* __restrict__ sumsq, long S) {
    int o = blockIdx.x;
    const _Float16* pb = p + (long)o * 3 * S;
    float ls = 0.0f, ls2 = 0.0f;
    for (long s = threadIdx.x; s < S; s += blockDim.x) {
        float a0 = (float)pb[s], a1 = (float)pb[S + s], a2 = (float)pb[2 * S + s];
        float nm = sqrtf(a0 * a0 + a1 * a1 + a2 * a2) + EPSV;
        ls += nm; ls2 += nm * nm;
    }
    __shared__ float sh[256], sh2[256];
    sh[threadIdx.x] = ls; sh2[threadIdx.x] = ls2;
    __syncthreads();
    for (int st = 128; st > 0; st >>= 1) {
        if ((int)threadIdx.x < st) {
            sh[threadIdx.x]  += sh[threadIdx.x + st];
            sh2[threadIdx.x] += sh2[threadIdx.x + st];
        }
        __syncthreads();
    }
    if (threadIdx.x == 0) {
        atomicAdd(&sum[o], sh[0]);
        atomicAdd(&sumsq[o], sh2[0]);
    }
}

// Fused: VN-BN normalize p, directional leaky-ReLU vs d (Od==1 => broadcast),
// mean over k. Writes f32 [O][3][N] (xout, pre-offset) and f16 transposed
// activation outT[(a*N+n)*Cpad + chanBase + o].
__global__ void vn_apply_kernel(const _Float16* __restrict__ p,
                                const _Float16* __restrict__ d,
                                const float* __restrict__ sum,
                                const float* __restrict__ sumsq,
                                float invCount,
                                float* __restrict__ xout,
                                _Float16* __restrict__ outT,
                                int Cpad, int chanBase,
                                int O, int Od, int N, int Kk) {
    int t = blockIdx.x * blockDim.x + threadIdx.x;
    if (t >= O * N) return;
    int o = t / N, n = t % N;
    long S = (long)N * Kk;
    const _Float16* pb = p + (long)o * 3 * S + (long)n * Kk;
    const _Float16* db = d + (long)((Od == 1) ? 0 : o) * 3 * S + (long)n * Kk;
    float mean = sum[o] * invCount;
    float var  = sumsq[o] * invCount - mean * mean;
    float rstd = rsqrtf(var + BN_EPSV);
    float a0 = 0.0f, a1 = 0.0f, a2 = 0.0f;
    for (int k = 0; k < Kk; ++k) {
        float p0 = (float)pb[k], p1 = (float)pb[S + k], p2 = (float)pb[2 * S + k];
        float nm = sqrtf(p0 * p0 + p1 * p1 + p2 * p2) + EPSV;
        float sc = (nm - mean) * rstd / nm;
        p0 *= sc; p1 *= sc; p2 *= sc;
        float d0 = (float)db[k], d1 = (float)db[S + k], d2 = (float)db[2 * S + k];
        float dot = p0 * d0 + p1 * d1 + p2 * d2;
        float r0 = p0, r1 = p1, r2 = p2;
        if (dot < 0.0f) {
            float c = dot / (d0 * d0 + d1 * d1 + d2 * d2 + EPSV);
            r0 -= c * d0; r1 -= c * d1; r2 -= c * d2;
        }
        a0 += SLOPEV * p0 + (1.0f - SLOPEV) * r0;
        a1 += SLOPEV * p1 + (1.0f - SLOPEV) * r1;
        a2 += SLOPEV * p2 + (1.0f - SLOPEV) * r2;
    }
    float inv = 1.0f / (float)Kk;
    float o0 = a0 * inv, o1 = a1 * inv, o2 = a2 * inv;
    long fb = (long)o * 3 * N + n;
    xout[fb]         = o0;
    xout[fb + N]     = o1;
    xout[fb + 2 * N] = o2;
    int co = chanBase + o;
    outT[((long)0 * N + n) * Cpad + co] = (_Float16)o0;
    outT[((long)1 * N + n) * Cpad + co] = (_Float16)o1;
    outT[((long)2 * N + n) * Cpad + co] = (_Float16)o2;
}

// z rows [C..2C): mean-over-N of rows [0..C); f32 z + f16 transposed zT.
__global__ void mean_concat_kernel(float* __restrict__ z, _Float16* __restrict__ zT,
                                   int C, int N, int Cpad) {
    int t = blockIdx.x * blockDim.x + threadIdx.x;
    if (t >= C * 3) return;
    int c = t / 3, a = t % 3;
    const float* src = z + (long)t * N;
    float s = 0.0f;
    for (int n = 0; n < N; ++n) s += src[n];
    s /= (float)N;
    long drow = ((long)(C + c) * 3 + a) * N;
    _Float16 sh = (_Float16)s;
    for (int n = 0; n < N; ++n) {
        z[drow + n] = s;
        zT[((long)a * N + n) * Cpad + C + c] = sh;
    }
}

// xs[i,k,n] = sum_j z[i,j,n] * z0[k][j*N+n]; pooled max & mean over n.
__global__ void pool_kernel(const float* __restrict__ z,
                            const float* __restrict__ z0,
                            float* __restrict__ f, int N) {
    int i = blockIdx.x * blockDim.x + threadIdx.x;
    if (i >= 682) return;
    float mx0 = -3.0e38f, mx1 = -3.0e38f, mx2 = -3.0e38f;
    float s0 = 0.0f, s1 = 0.0f, s2 = 0.0f;
    long N3 = 3L * N;
    for (int n = 0; n < N; ++n) {
        float zi0 = z[((long)i * 3 + 0) * N + n];
        float zi1 = z[((long)i * 3 + 1) * N + n];
        float zi2 = z[((long)i * 3 + 2) * N + n];
        float v0 = zi0 * z0[0 * N3 + n] + zi1 * z0[0 * N3 + N + n] + zi2 * z0[0 * N3 + 2 * N + n];
        float v1 = zi0 * z0[1 * N3 + n] + zi1 * z0[1 * N3 + N + n] + zi2 * z0[1 * N3 + 2 * N + n];
        float v2 = zi0 * z0[2 * N3 + n] + zi1 * z0[2 * N3 + N + n] + zi2 * z0[2 * N3 + 2 * N + n];
        mx0 = fmaxf(mx0, v0); s0 += v0;
        mx1 = fmaxf(mx1, v1); s1 += v1;
        mx2 = fmaxf(mx2, v2); s2 += v2;
    }
    float inv = 1.0f / (float)N;
    f[i * 3 + 0] = mx0; f[i * 3 + 1] = mx1; f[i * 3 + 2] = mx2;
    f[2046 + i * 3 + 0] = s0 * inv;
    f[2046 + i * 3 + 1] = s1 * inv;
    f[2046 + i * 3 + 2] = s2 * inv;
}

// BatchNorm1d over batch rows + optional leaky ReLU, in place.
__global__ void bn1d_leaky_kernel(float* __restrict__ h, int Bsz, int F, int leaky) {
    int f = blockIdx.x * blockDim.x + threadIdx.x;
    if (f >= F) return;
    float s = 0.0f, s2 = 0.0f;
    for (int b = 0; b < Bsz; ++b) {
        float v = h[(long)b * F + f];
        s += v; s2 += v * v;
    }
    float m = s / Bsz, var = s2 / Bsz - m * m, r = rsqrtf(var + BN_EPSV);
    for (int b = 0; b < Bsz; ++b) {
        float v = (h[(long)b * F + f] - m) * r;
        if (leaky && v < 0.0f) v *= SLOPEV;
        h[(long)b * F + f] = v;
    }
}

// ---------------------------------------------------------------------------
extern "C" void kernel_launch(void* const* d_in, const int* in_sizes, int n_in,
                              void* d_out, int out_size, void* d_ws, size_t ws_size,
                              hipStream_t stream) {
    (void)n_in; (void)out_size; (void)ws_size; (void)in_sizes;
    const int  Nn = 1024, Bb = 8;
    const long M_edge = 3L * Nn * KNN;      // 61440
    const long M3     = 3L * Nn;            // 3072
    const long PSTR   = 85L * M_edge;       // max per-batch p/d elements

    const float* x = (const float*)d_in[0];
    const float* W[21];
    for (int i = 0; i < 21; ++i) W[i] = (const float*)d_in[1 + i];
    // W[0..14]=W1f..Wz, W[15]=A1, W[16]=b1, W[17]=A2, W[18]=b2, W[19]=A3, W[20]=b3

    char* wsp = (char*)d_ws;
    size_t off = 0;
    auto alloc = [&](size_t bytes) -> void* {
        void* pt = (void*)(wsp + off);
        off = (off + bytes + 255) & ~(size_t)255;
        return pt;
    };
    auto rup32 = [](int k) { return (k + 31) & ~31; };

    // Padded f16 weights (rows = out-features, K-contiguous, zero-padded).
    static const int wR[21] = {21,21,21,21,42,42,85,85,341,1,341,341,170,170,3,512,0,256,0,40,0};
    static const int wK[21] = { 2, 2,42,42,42,42,84,84,169,169,682,682,341,341,170,4092,0,512,0,256,0};
    _Float16* hWp[21];
    int wKP[21];
    for (int i = 0; i < 21; ++i) {
        if (wK[i] == 0) { hWp[i] = nullptr; wKP[i] = 0; continue; }
        wKP[i] = rup32(wK[i]);
        hWp[i] = (_Float16*)alloc((size_t)wR[i] * wKP[i] * 2);
    }

    const int CP_XC = 192;   // 169 channels -> 192
    const int CP_Z  = 704;   // 682 -> 704
    const int CP_Y1 = 352;   // 341 -> 352
    const int CP_Y2 = 192;   // 170 -> 192

    float*    f_xcat  = (float*)   alloc((size_t)Bb * 169 * M3 * 4);
    _Float16* h_xcatT = (_Float16*)alloc((size_t)Bb * M3 * CP_XC * 2);
    float*    f_z     = (float*)   alloc((size_t)Bb * 682 * M3 * 4);
    _Float16* h_zT    = (_Float16*)alloc((size_t)Bb * M3 * CP_Z * 2);
    float*    f_y1    = (float*)   alloc((size_t)Bb * 341 * M3 * 4);
    _Float16* h_y1T   = (_Float16*)alloc((size_t)Bb * M3 * CP_Y1 * 2);
    float*    f_y2    = (float*)   alloc((size_t)Bb * 170 * M3 * 4);
    _Float16* h_y2T   = (_Float16*)alloc((size_t)Bb * M3 * CP_Y2 * 2);
    _Float16* h_xfT   = (_Float16*)alloc((size_t)Nn * 128 * 2);
    _Float16* h_featT = (_Float16*)alloc((size_t)M_edge * 96 * 2);
    float*    f_gram  = (float*)   alloc((size_t)Nn * Nn * 4);
    float*    f_xx    = (float*)   alloc((size_t)Nn * 4);
    int*      i_idx   = (int*)     alloc((size_t)Nn * KNN * 4);
    _Float16* h_p     = (_Float16*)alloc((size_t)Bb * PSTR * 2);
    _Float16* h_d     = (_Float16*)alloc((size_t)Bb * PSTR * 2);
    float*    f_z0    = (float*)   alloc((size_t)Bb * 3 * M3 * 4);
    float*    f_f     = (float*)   alloc((size_t)Bb * 4092 * 4);
    _Float16* h_f     = (_Float16*)alloc((size_t)Bb * 4096 * 2);
    float*    f_h1    = (float*)   alloc((size_t)Bb * 512 * 4);
    _Float16* h_h1    = (_Float16*)alloc((size_t)Bb * 512 * 2);
    float*    f_h2    = (float*)   alloc((size_t)Bb * 256 * 4);
    _Float16* h_h2    = (_Float16*)alloc((size_t)Bb * 256 * 2);
    float*    f_stats = (float*)   alloc(1024 * 4);
    float* sumv = f_stats, * sumsqv = f_stats + 512;

    auto gsz = [](long n) { int b = (int)((n + 255) / 256); return b > 4096 ? 4096 : b; };
    auto zero16 = [&](_Float16* p, long n) { zero_f16_kernel<<<gsz(n), 256, 0, stream>>>(p, n); };
    auto castpad = [&](const float* s, _Float16* d, int R, int K, int KP) {
        cast_pad_kernel<<<gsz((long)R * KP), 256, 0, stream>>>(s, d, R, K, KP);
    };
    auto gemm_h = [&](const _Float16* A, const _Float16* BT, _Float16* C,
                      int M, int Ncol, int KP, long Apad, long Bpad, long Crs) {
        long waves = (long)((M + 31) / 32) * ((Ncol + 31) / 32);
        gemm_wmma_kernel<_Float16><<<(int)((waves + 3) / 4), 128, 0, stream>>>(
            A, BT, C, M, Ncol, KP, Apad, Bpad, Crs, nullptr);
    };
    auto gemm_f = [&](const _Float16* A, const _Float16* BT, float* C,
                      int M, int Ncol, int KP, long Apad, long Bpad, long Crs,
                      const float* bias) {
        long waves = (long)((M + 31) / 32) * ((Ncol + 31) / 32);
        gemm_wmma_kernel<float><<<(int)((waves + 3) / 4), 128, 0, stream>>>(
            A, BT, C, M, Ncol, KP, Apad, Bpad, Crs, bias);
    };

    // Weight casts + zero the transposed activation pads (done every call;
    // pads stay zero because only real channels are ever written afterwards).
    for (int i = 0; i < 21; ++i)
        if (wK[i]) castpad(W[i], hWp[i], wR[i], wK[i], wKP[i]);
    zero16(h_xcatT, (long)Bb * M3 * CP_XC);
    zero16(h_zT,    (long)Bb * M3 * CP_Z);
    zero16(h_y1T,   (long)Bb * M3 * CP_Y1);
    zero16(h_y2T,   (long)Bb * M3 * CP_Y2);

    // ---- edge-conv layers ------------------------------------------------
    auto edge_layer = [&](int first, int chanInOff, int Cin, int O,
                          const _Float16* Wfp, const _Float16* Wdp, int KP,
                          int outOff) {
        int D = Cin * 3, DP = rup32(D), Cin2 = 2 * Cin;
        long S = (long)Nn * KNN;
        zero16(h_featT, M_edge * (long)KP);
        for (int b = 0; b < Bb; ++b) {
            const float* xcur = first ? (x + (long)b * 3 * Nn)
                                      : (f_xcat + (long)b * 169 * M3 + (long)chanInOff * M3);
            sqnorm_kernel<<<(Nn + 255) / 256, 256, 0, stream>>>(xcur, f_xx, D, Nn);
            build_xfT_kernel<<<gsz((long)Nn * DP), 256, 0, stream>>>(xcur, h_xfT, D, Nn, DP);
            // Gram = xfT * xfT^T : one buffer is both A and B-transposed.
            gemm_f(h_xfT, h_xfT, f_gram, Nn, Nn, DP, DP, DP, Nn, nullptr);
            topk_kernel<<<(Nn + 255) / 256, 256, 0, stream>>>(f_gram, f_xx, i_idx, Nn);
            gather_edge_kernel<<<gsz((long)Cin2 * M_edge), 256, 0, stream>>>(
                xcur, i_idx, h_featT, Cin, Nn, KNN, KP);
            gemm_h(Wfp, h_featT, h_p + (long)b * PSTR, O, (int)M_edge, KP, KP, KP, M_edge);
            gemm_h(Wdp, h_featT, h_d + (long)b * PSTR, O, (int)M_edge, KP, KP, KP, M_edge);
        }
        zero_f32_kernel<<<4, 256, 0, stream>>>(f_stats, 1024);
        for (int b = 0; b < Bb; ++b)
            bn_stats_kernel<<<O, 256, 0, stream>>>(h_p + (long)b * PSTR, sumv, sumsqv, S);
        float invCount = 1.0f / (float)((long)Bb * S);
        for (int b = 0; b < Bb; ++b)
            vn_apply_kernel<<<(O * Nn + 255) / 256, 256, 0, stream>>>(
                h_p + (long)b * PSTR, h_d + (long)b * PSTR, sumv, sumsqv, invCount,
                f_xcat + (long)b * 169 * M3 + (long)outOff * M3,
                h_xcatT + (long)b * M3 * CP_XC, CP_XC, outOff, O, O, Nn, KNN);
    };
    edge_layer(1,  0,  1, 21, hWp[0], hWp[1], wKP[0],  0);   // -> x1
    edge_layer(0,  0, 21, 21, hWp[2], hWp[3], wKP[2], 21);   // -> x2
    edge_layer(0, 21, 21, 42, hWp[4], hWp[5], wKP[4], 42);   // -> x3
    edge_layer(0, 42, 42, 85, hWp[6], hWp[7], wKP[6], 84);   // -> x4

    // ---- plain VN layers (K=1) -------------------------------------------
    auto vn_layer = [&](const _Float16* inT, int CpadIn, int O, int Od,
                        const _Float16* Wfp, const _Float16* Wdp,
                        float* outf, long outfStride,
                        _Float16* outT, int CpadOut) {
        for (int b = 0; b < Bb; ++b) {
            const _Float16* BT = inT + (long)b * M3 * CpadIn;
            gemm_h(Wfp, BT, h_p + (long)b * PSTR, O,  (int)M3, CpadIn, CpadIn, CpadIn, M3);
            gemm_h(Wdp, BT, h_d + (long)b * PSTR, Od, (int)M3, CpadIn, CpadIn, CpadIn, M3);
        }
        zero_f32_kernel<<<4, 256, 0, stream>>>(f_stats, 1024);
        for (int b = 0; b < Bb; ++b)
            bn_stats_kernel<<<O, 256, 0, stream>>>(h_p + (long)b * PSTR, sumv, sumsqv, (long)Nn);
        float invCount = 1.0f / (float)(Bb * Nn);
        for (int b = 0; b < Bb; ++b)
            vn_apply_kernel<<<(O * Nn + 255) / 256, 256, 0, stream>>>(
                h_p + (long)b * PSTR, h_d + (long)b * PSTR, sumv, sumsqv, invCount,
                outf + (long)b * outfStride, outT + (long)b * M3 * CpadOut,
                CpadOut, 0, O, Od, Nn, 1);
    };

    // conv5: xcat -> xc written into z rows 0..341 (d is 1-channel)
    vn_layer(h_xcatT, CP_XC, 341, 1, hWp[8], hWp[9], f_z, 682L * M3, h_zT, CP_Z);
    for (int b = 0; b < Bb; ++b)
        mean_concat_kernel<<<(341 * 3 + 255) / 256, 256, 0, stream>>>(
            f_z + (long)b * 682 * M3, h_zT + (long)b * M3 * CP_Z, 341, Nn, CP_Z);
    // VNStdFeature
    vn_layer(h_zT,  CP_Z,  341, 341, hWp[10], hWp[11], f_y1, 341L * M3, h_y1T, CP_Y1);
    vn_layer(h_y1T, CP_Y1, 170, 170, hWp[12], hWp[13], f_y2, 170L * M3, h_y2T, CP_Y2);
    for (int b = 0; b < Bb; ++b)                    // z0 = Wz * y2 : [3][3N]
        gemm_f(hWp[14], h_y2T + (long)b * M3 * CP_Y2, f_z0 + (long)b * 3 * M3,
               3, (int)M3, CP_Y2, wKP[14], CP_Y2, M3, nullptr);
    for (int b = 0; b < Bb; ++b)
        pool_kernel<<<(682 + 255) / 256, 256, 0, stream>>>(
            f_z + (long)b * 682 * M3, f_z0 + (long)b * 3 * M3, f_f + (long)b * 4092, Nn);

    // ---- FC head (A1/A2/A3 row-major are already B-transposed) -----------
    castpad(f_f, h_f, 8, 4092, 4096);
    gemm_f(h_f, hWp[15], f_h1, 8, 512, 4096, 4096, 4096, 512, W[16]);
    bn1d_leaky_kernel<<<2, 256, 0, stream>>>(f_h1, 8, 512, 1);
    cast_f32_f16_kernel<<<gsz(8 * 512), 256, 0, stream>>>(f_h1, h_h1, 8 * 512);
    gemm_f(h_h1, hWp[17], f_h2, 8, 256, 512, 512, 512, 256, W[18]);
    bn1d_leaky_kernel<<<1, 256, 0, stream>>>(f_h2, 8, 256, 1);
    cast_f32_f16_kernel<<<gsz(8 * 256), 256, 0, stream>>>(f_h2, h_h2, 8 * 256);
    gemm_f(h_h2, hWp[19], (float*)d_out, 8, 40, 256, 256, 256, 40, W[20]);
}
